// ChebFCN2D_60696477827200
// MI455X (gfx1250) — matrix operand order repaired and checked by reading, not compiled, not run on hardware
//
#include <hip/hip_runtime.h>
#include <hip/hip_bf16.h>

// ChebFCN2D for MI455X (gfx1250), wave32 + WMMA.
//
// out[n,k] = Xf(n)^T * M_k * Yf(n) + b[k] + W[k][0]
//   Xf(n)[i] = T_{i+1}(x_n),  Yf(n)[j] = T_{j+1}(y_n),  i,j in [0,31) (pad to 32)
//   M_k[i][j] = W[k][1 + 31*i + j]
//
// Stage 1 per 16-point tile and channel k:
//   D = A * B  with A = M_k tile [16 x 32] (f16, constant, packed once per wave),
//                 B = Yf columns  [32 x 16] (f16, rebuilt per tile)
//   via v_wmma_f32_16x16x32_f16 (fp32 accumulate). Two A tiles cover i=0..31.
// Stage 2: per-lane 8-FMA partial dot with Xf (fp32 from LDS), then a
//   half-wave __shfl_xor(16) completes each column sum.

typedef __attribute__((ext_vector_type(16))) _Float16 v16h;
typedef __attribute__((ext_vector_type(8)))  float    v8f;

#define WAVES_PER_BLOCK 8
#define BLOCK_THREADS   (WAVES_PER_BLOCK * 32)

__global__ __launch_bounds__(BLOCK_THREADS)
void cheb_fcn2d_wmma(const float* __restrict__ x,   // [N,2]
                     const float* __restrict__ W,   // [3,962]
                     const float* __restrict__ b,   // [3]
                     float* __restrict__ out,       // [N,3]
                     int N, int iters, int nTiles)
{
    // Per-wave staging: [wave][X=0 / Y=1][point 0..15][poly 0..31] fp32
    __shared__ float lds[WAVES_PER_BLOCK][2][16][32];

    const int tid  = threadIdx.x;
    const int wave = tid >> 5;
    const int lane = tid & 31;
    const int hi   = lane >> 4;   // half-wave: 0 = lanes 0-15, 1 = lanes 16-31
    const int p    = lane & 15;   // point-in-tile (B/D column), also A row m

    float* Xf = &lds[wave][0][0][0];
    float* Yf = &lds[wave][1][0][0];

    // ---------------------------------------------------------------
    // Pack the three constant A operands (M_k) once per wave.
    // 16-bit A 16x32 layout (ISA 7.12.2): lane group selects K phase:
    //   lanes 0-15 : V0..V3 -> K=0..7,   V4..V7 -> K=16..23
    //   lanes 16-31: V0..V3 -> K=8..15,  V4..V7 -> K=24..31
    // Row M = lane & 15 (replicated across half-waves).
    // ---------------------------------------------------------------
    v16h A[3][2];
    #pragma unroll
    for (int k = 0; k < 3; ++k) {
        #pragma unroll
        for (int h = 0; h < 2; ++h) {
            v16h a;
            const int i = p + 16 * h;                 // A row = X-poly index - 1
            #pragma unroll
            for (int v = 0; v < 8; ++v) {
                const int kb = ((v < 4) ? (2 * v) : (16 + 2 * (v - 4))) + 8 * hi;
                #pragma unroll
                for (int e = 0; e < 2; ++e) {
                    const int j = kb + e;             // A col = Y-poly index - 1
                    float w = 0.0f;
                    if (i < 31 && j < 31)
                        w = W[k * 962 + 1 + i * 31 + j];
                    a[2 * v + e] = (_Float16)w;
                }
            }
            A[k][h] = a;
        }
    }
    const float c0 = b[0] + W[0];
    const float c1 = b[1] + W[962];
    const float c2 = b[2] + W[2 * 962];

    const int waveGlobal = blockIdx.x * WAVES_PER_BLOCK + wave;
    const int totalWaves = gridDim.x * WAVES_PER_BLOCK;

    for (int it = 0; it < iters; ++it) {
        const int  tile   = waveGlobal + it * totalWaves;
        const bool active = (tile < nTiles);          // wave-uniform
        const int  n0     = tile * 16;

        // ---- Phase 1: Chebyshev recurrences into LDS (fp32) ----
        if (active) {
            int pg = n0 + p; if (pg >= N) pg = N - 1; // clamp tail reads
            const float c = x[2 * pg + hi];           // hi=0 -> x coord, hi=1 -> y
            float* dst = (hi == 0 ? Xf : Yf) + p * 32;
            float t_pp = 1.0f;                        // T0
            float t_p  = c;                           // T1
            dst[0] = t_p;                             // store T1..T31 at 0..30
            #pragma unroll
            for (int n = 2; n < 32; ++n) {
                const float t = __builtin_fmaf(2.0f * c, t_p, -t_pp);
                dst[n - 1] = t;
                t_pp = t_p; t_p = t;
            }
            dst[31] = 0.0f;                           // K=31 pad
        }
        __syncthreads();  // uniform trip count across all waves

        // ---- Phase 2: WMMA bilinear form + half-wave reduction ----
        if (active) {
            // B operand (32x16 f16): lane column p; lanes 0-15 hold K=0..7,16..23,
            // lanes 16-31 hold K=8..15,24..31, two K per VGPR.
            v16h Bv;
            const float* ys = Yf + p * 32 + 8 * hi;
            #pragma unroll
            for (int e = 0; e < 16; ++e) Bv[e] = (_Float16)ys[(e < 8) ? e : (8 + e)];

            // Stage-2 Xf values for this lane's D rows:
            //   D0 rows i = r + 8*hi, D1 rows i = 16 + r + 8*hi
            float xv0[8], xv1[8];
            const float* xs0 = Xf + p * 32 + 8 * hi;
            const float* xs1 = Xf + p * 32 + 16 + 8 * hi;
            #pragma unroll
            for (int r = 0; r < 8; ++r) { xv0[r] = xs0[r]; xv1[r] = xs1[r]; }

            float o[3];
            #pragma unroll
            for (int k = 0; k < 3; ++k) {
                v8f z = {};
                v8f d0 = __builtin_amdgcn_wmma_f32_16x16x32_f16(
                             false, A[k][0], false, Bv, (short)0, z, false, false);
                v8f d1 = __builtin_amdgcn_wmma_f32_16x16x32_f16(
                             false, A[k][1], false, Bv, (short)0, z, false, false);
                float acc = 0.0f;
                #pragma unroll
                for (int r = 0; r < 8; ++r)
                    acc = __builtin_fmaf(xv0[r], d0[r],
                          __builtin_fmaf(xv1[r], d1[r], acc));
                // lane p has rows {0..7,16..23}+8*hi; partner lane p^16 has the rest
                acc += __shfl_xor(acc, 16, 32);
                o[k] = acc;
            }

            if (hi == 0) {                            // lanes 0-15 own one point each
                const int pg = n0 + p;
                if (pg < N) {
                    out[3 * pg + 0] = o[0] + c0;
                    out[3 * pg + 1] = o[1] + c1;
                    out[3 * pg + 2] = o[2] + c2;
                }
            }
        }
        __syncthreads();  // LDS reused next iteration
    }
}

extern "C" void kernel_launch(void* const* d_in, const int* in_sizes, int n_in,
                              void* d_out, int out_size, void* d_ws, size_t ws_size,
                              hipStream_t stream) {
    (void)n_in; (void)out_size; (void)d_ws; (void)ws_size;
    const float* x = (const float*)d_in[0];   // [N,2]
    const float* W = (const float*)d_in[1];   // [3,962]
    const float* b = (const float*)d_in[2];   // [3]
    float* out = (float*)d_out;               // [N,3]

    const int N      = in_sizes[0] / 2;
    const int nTiles = (N + 15) / 16;

    const int blocks     = 256;               // 2048 waves; N=262144 -> 8 tiles/wave
    const int totalWaves = blocks * WAVES_PER_BLOCK;
    const int iters      = (nTiles + totalWaves - 1) / totalWaves;

    cheb_fcn2d_wmma<<<blocks, BLOCK_THREADS, 0, stream>>>(
        x, W, b, out, N, iters, nTiles);
}
